// LeWinBlock_884763263602
// MI455X (gfx1250) — matrix operand order repaired and checked by reading, not compiled, not run on hardware
//
#include <hip/hip_runtime.h>
#include <hip/hip_bf16.h>
#include <math.h>

typedef _Float16 v16h __attribute__((ext_vector_type(16)));
typedef _Float16 v8h  __attribute__((ext_vector_type(8)));
typedef float    v8f  __attribute__((ext_vector_type(8)));

#define NPIX   36864      // 192*192
#define IMGW   192
#define CDIM   128
#define INNERD 512
#define HIDD   64

static __device__ inline v8f wmma32(v16h a, v16h b, v8f c) {
  return __builtin_amdgcn_wmma_f32_16x16x32_f16(false, a, false, b, (short)0, c,
                                                false, false);
}

static __device__ inline v16h zero16() {
  v16h z;
#pragma unroll
  for (int i = 0; i < 16; ++i) z[i] = (_Float16)0.0f;
  return z;
}

// A fragment (16x32) from f16 row-major matrix, per-lane = two contiguous 16B
// chunks at k = k0+16*half+[0..7] and k0+16*half+8+[0..7].
static __device__ inline v16h load_a16(const _Float16* __restrict__ Wp, int ldk,
                                       int row0, int k0) {
  int m  = row0 + (threadIdx.x & 15);
  int h8 = ((threadIdx.x >> 4) & 1) * 16;
  const _Float16* base = Wp + (size_t)m * ldk + k0 + h8;
  v8h lo = *(const v8h*)(base);
  v8h hi = *(const v8h*)(base + 8);
  v16h r;
#pragma unroll
  for (int i = 0; i < 8; ++i) { r[i] = lo[i]; r[i + 8] = hi[i]; }
  return r;
}

// B fragment (32x16) from a K-contiguous ("pixel-major") matrix: per-lane 16
// contiguous halves at row n, k = k0 + 16*half + [0..15]. One 32B load.
static __device__ inline v16h load_b16(const _Float16* __restrict__ Xp, int ldk,
                                       int n, int k0) {
  int h16 = ((threadIdx.x >> 4) & 1) * 16;
  return *(const v16h*)(Xp + (size_t)n * ldk + k0 + h16);
}

// ------------------------- weight conversion ---------------------------------
__global__ __launch_bounds__(256) void cvt_f16(const float* __restrict__ src,
                                               _Float16* __restrict__ dst, int n) {
  int i = blockIdx.x * 256 + threadIdx.x;
  if (i < n) dst[i] = (_Float16)src[i];
}
// w2 [O][K][3][3] -> [tap][O][K] so conv A-fragments are contiguous
__global__ __launch_bounds__(256) void cvt_w2(const float* __restrict__ src,
                                              _Float16* __restrict__ dst) {
  int i = blockIdx.x * 256 + threadIdx.x;
  if (i >= 512 * 512 * 9) return;
  int o = i / (512 * 9);
  int rem = i % (512 * 9);
  int k = rem / 9, t = rem % 9;
  dst[(size_t)t * 512 * 512 + (size_t)o * 512 + k] = (_Float16)src[i];
}

// -------- channel LayerNorm (x is f32 [C][N]); xn written pixel-major [N][C] -
__global__ __launch_bounds__(256) void ln_kernel(const float* __restrict__ x,
                                                 const float* __restrict__ g,
                                                 const float* __restrict__ b,
                                                 _Float16* __restrict__ xn_pm) {
  int p = blockIdx.x * 256 + threadIdx.x;
  if (p >= NPIX) return;
  float s = 0.f, ss = 0.f;
  for (int c = 0; c < CDIM; ++c) {
    float v = x[(size_t)c * NPIX + p];
    s += v; ss += v * v;
  }
  float mu   = s * (1.0f / CDIM);
  float var  = ss * (1.0f / CDIM) - mu * mu;
  float rstd = rsqrtf(var + 1e-5f);
  for (int c = 0; c < CDIM; ++c) {
    float v = (x[(size_t)c * NPIX + p] - mu) * rstd * g[c] + b[c];
    xn_pm[(size_t)p * CDIM + c] = (_Float16)v;
  }
}

// ---------------- WMMA GEMM: out[O][N] = W[O][K] @ X^T, X pixel-major [N][K] -
// Wave tile 32x64 (2 M-tiles x 4 N-subtiles, B reused across both M-tiles).
// Block tile 128x64 (4 waves). Grid: (N/64, O/128).
// mode 0: out_pm f16 [N][O] (+bias)        (vectorized b128 stores)
// mode 1: out_cm f16 [O][N] (+bias)        (q/kv projections)
// mode 2: out_acc f32 [O][N] += (+bias)    (residual adds)
__global__ __launch_bounds__(128) void gemm_wmma(
    const _Float16* __restrict__ Wh, const _Float16* __restrict__ Xpm,
    const float* __restrict__ bias, _Float16* __restrict__ out_pm,
    _Float16* __restrict__ out_cm, float* __restrict__ out_acc,
    int O, int K, int N, int mode) {
  int wave = threadIdx.x >> 5;
  int m0   = blockIdx.y * 128 + wave * 32;
  int n0   = blockIdx.x * 64;
  int ln   = threadIdx.x & 15;
  v8f acc[2][4];
#pragma unroll
  for (int mt = 0; mt < 2; ++mt)
#pragma unroll
    for (int s = 0; s < 4; ++s) acc[mt][s] = (v8f){0, 0, 0, 0, 0, 0, 0, 0};
  for (int k0 = 0; k0 < K; k0 += 32) {
    v16h a0 = load_a16(Wh, K, m0, k0);
    v16h a1 = load_a16(Wh, K, m0 + 16, k0);
#pragma unroll
    for (int s = 0; s < 4; ++s) {
      v16h b = load_b16(Xpm, K, n0 + s * 16 + ln, k0);
      acc[0][s] = wmma32(a0, b, acc[0][s]);
      acc[1][s] = wmma32(a1, b, acc[1][s]);
    }
  }
  int h8 = ((threadIdx.x >> 4) & 1) * 8;
  if (mode == 0) {
#pragma unroll
    for (int mt = 0; mt < 2; ++mt)
#pragma unroll
      for (int s = 0; s < 4; ++s) {
        v8h o;
#pragma unroll
        for (int r = 0; r < 8; ++r)
          o[r] = (_Float16)(acc[mt][s][r] +
                            (bias ? bias[m0 + mt * 16 + h8 + r] : 0.0f));
        *(v8h*)(out_pm + (size_t)(n0 + s * 16 + ln) * O + m0 + mt * 16 + h8) = o;
      }
  } else if (mode == 1) {
#pragma unroll
    for (int mt = 0; mt < 2; ++mt)
#pragma unroll
      for (int s = 0; s < 4; ++s)
#pragma unroll
        for (int r = 0; r < 8; ++r) {
          int m = m0 + mt * 16 + h8 + r;
          float v = acc[mt][s][r] + (bias ? bias[m] : 0.0f);
          out_cm[(size_t)m * N + n0 + s * 16 + ln] = (_Float16)v;
        }
  } else {
#pragma unroll
    for (int mt = 0; mt < 2; ++mt)
#pragma unroll
      for (int s = 0; s < 4; ++s)
#pragma unroll
        for (int r = 0; r < 8; ++r) {
          int m = m0 + mt * 16 + h8 + r;
          float v = acc[mt][s][r] + (bias ? bias[m] : 0.0f);
          out_acc[(size_t)m * N + n0 + s * 16 + ln] += v;
        }
  }
}

// -------- 3x3 conv 512->512 implicit GEMM, X/out pixel-major, bias+GELU ------
// Same 32x64 wave tile / 128x64 block tile as gemm_wmma. Grid: (N/64, 4).
__global__ __launch_bounds__(128) void conv3x3_gelu(
    const _Float16* __restrict__ Wh /*[9][O][K]*/, const _Float16* __restrict__ Xpm,
    const float* __restrict__ bias, _Float16* __restrict__ out_pm) {
  const int K = 512, O = 512;
  int wave = threadIdx.x >> 5;
  int m0   = blockIdx.y * 128 + wave * 32;
  int n0   = blockIdx.x * 64;
  int ln   = threadIdx.x & 15;
  v8f acc[2][4];
#pragma unroll
  for (int mt = 0; mt < 2; ++mt)
#pragma unroll
    for (int s = 0; s < 4; ++s) acc[mt][s] = (v8f){0, 0, 0, 0, 0, 0, 0, 0};
  int hh[4], ww[4];
#pragma unroll
  for (int s = 0; s < 4; ++s) {
    int np = n0 + s * 16 + ln;
    hh[s] = np / IMGW; ww[s] = np % IMGW;
  }
  for (int t = 0; t < 9; ++t) {
    int dy = t / 3 - 1, dx = t % 3 - 1;
    const _Float16* Wt = Wh + (size_t)t * O * K;
    int col[4]; bool ok[4];
#pragma unroll
    for (int s = 0; s < 4; ++s) {
      int h2 = hh[s] + dy, w2 = ww[s] + dx;
      ok[s]  = ((unsigned)h2 < (unsigned)IMGW) && ((unsigned)w2 < (unsigned)IMGW);
      col[s] = h2 * IMGW + w2;
    }
    for (int k0 = 0; k0 < K; k0 += 32) {
      v16h a0 = load_a16(Wt, K, m0, k0);
      v16h a1 = load_a16(Wt, K, m0 + 16, k0);
#pragma unroll
      for (int s = 0; s < 4; ++s) {
        v16h b = ok[s] ? load_b16(Xpm, K, col[s], k0) : zero16();
        acc[0][s] = wmma32(a0, b, acc[0][s]);
        acc[1][s] = wmma32(a1, b, acc[1][s]);
      }
    }
  }
  int h8 = ((threadIdx.x >> 4) & 1) * 8;
#pragma unroll
  for (int mt = 0; mt < 2; ++mt)
#pragma unroll
    for (int s = 0; s < 4; ++s) {
      v8h o;
#pragma unroll
      for (int r = 0; r < 8; ++r) {
        float v = acc[mt][s][r] + bias[m0 + mt * 16 + h8 + r];
        o[r] = (_Float16)(0.5f * v * (1.0f + erff(v * 0.70710678118654752f)));
      }
      *(v8h*)(out_pm + (size_t)(n0 + s * 16 + ln) * O + m0 + mt * 16 + h8) = o;
    }
}

// ---------------- rotary (with the reference's exact quirks) -----------------
__global__ __launch_bounds__(256) void rotary_kernel(_Float16* __restrict__ q,
                                                     _Float16* __restrict__ kk) {
  size_t i = (size_t)blockIdx.x * 256 + threadIdx.x;
  if (i >= (size_t)INNERD * NPIX) return;
  int d      = (int)(i & 63);
  size_t rem = i % ((size_t)HIDD * NPIX);
  int pos    = (int)(rem >> 6);
  int a      = pos / IMGW, b = pos % IMGW;
  int jj     = d & 31;
  float t;
  if (jj < 16) {
    float coord = -1.0f + 2.0f * a / 191.0f;
    t = coord * (1.0f + 4.0f * jj / 15.0f) * 3.14159265358979f;
  } else {
    float coord = -1.0f + 2.0f * b / 191.0f;
    t = coord * (1.0f + 4.0f * (jj - 16) / 15.0f) * 3.14159265358979f;
  }
  float sn = __sinf(t), cs = __cosf(t);
  float qv = (float)q[i], kv = (float)kk[i];
  if (d & 1) {
    q[i]  = (_Float16)(qv * (cs + sn));
    kk[i] = (_Float16)(kv * (1.0f + sn) + cs);
  } else {
    q[i]  = (_Float16)(qv * (cs - sn));
    kk[i] = (_Float16)(kv * (1.0f - sn) + cs);
  }
}

// -------- window attention: one block per [256x64] head-window chunk ---------
// q/k/v channel-major flat (windows are contiguous 16384-elem chunks);
// output written pixel-major [N][512] for the following wo GEMM.
__global__ __launch_bounds__(256) void attn_kernel(
    const _Float16* __restrict__ q, const _Float16* __restrict__ kmat,
    const _Float16* __restrict__ v, _Float16* __restrict__ out_pm) {
  __shared__ _Float16 Kc[32 * 64];   // [key][ch]
  __shared__ _Float16 Vt[64 * 32];   // [ch][key]  (transposed for P@V B-frags)
  __shared__ _Float16 Ps[8][16 * 32];
  const float SC = 0.125f;  // 64^-0.5
  size_t wbase = (size_t)blockIdx.x * (256 * 64);
  int wave = threadIdx.x >> 5;
  int lane = threadIdx.x & 31;
  int half = lane >> 4;
  int ln   = lane & 15;

  // Q fragments: rows [wave*32, wave*32+32), K=64 in two 32-steps (b128 loads)
  v16h qa[2][2];
#pragma unroll
  for (int mt = 0; mt < 2; ++mt)
#pragma unroll
    for (int ks = 0; ks < 2; ++ks)
      qa[mt][ks] = load_a16(q + wbase, 64, wave * 32 + mt * 16, ks * 32);

  float mrun[2][8], lrun[2][8];
  v8f oacc[2][4];
#pragma unroll
  for (int mt = 0; mt < 2; ++mt) {
#pragma unroll
    for (int r = 0; r < 8; ++r) { mrun[mt][r] = -1e30f; lrun[mt][r] = 0.0f; }
#pragma unroll
    for (int s = 0; s < 4; ++s) oacc[mt][s] = (v8f){0, 0, 0, 0, 0, 0, 0, 0};
  }

  for (int c = 0; c < 8; ++c) {   // 8 chunks of 32 keys
    size_t cbase = wbase + (size_t)(c * 32) * 64;
    {
      int idx = threadIdx.x;                       // 256 threads x 8 halves
      v8h k8 = *(const v8h*)(kmat + cbase + idx * 8);
      *(v8h*)(Kc + idx * 8) = k8;                  // contiguous
      v8h v8 = *(const v8h*)(v + cbase + idx * 8);
      int key = idx >> 3;
      int ch0 = (idx & 7) * 8;
#pragma unroll
      for (int e = 0; e < 8; ++e) Vt[(ch0 + e) * 32 + key] = v8[e];
    }
    __syncthreads();

#pragma unroll
    for (int mt = 0; mt < 2; ++mt) {
      v8f S[2];
#pragma unroll
      for (int ns = 0; ns < 2; ++ns) {
        v8f sacc = (v8f){0, 0, 0, 0, 0, 0, 0, 0};
        int key = ns * 16 + ln;
#pragma unroll
        for (int ks = 0; ks < 2; ++ks) {
          v16h b = *(const v16h*)(Kc + key * 64 + ks * 32 + 16 * half);
          sacc = wmma32(qa[mt][ks], b, sacc);
        }
        S[ns] = sacc;
      }
      float p0a[8], p1a[8];
#pragma unroll
      for (int r = 0; r < 8; ++r) {
        float s0 = S[0][r] * SC, s1 = S[1][r] * SC;
        float rm = fmaxf(s0, s1);
#pragma unroll
        for (int off = 1; off < 16; off <<= 1)
          rm = fmaxf(rm, __shfl_xor(rm, off, 32));
        float mnew = fmaxf(mrun[mt][r], rm);
        float scal = __expf(mrun[mt][r] - mnew);
        float p0 = __expf(s0 - mnew), p1 = __expf(s1 - mnew);
        float rs = p0 + p1;
#pragma unroll
        for (int off = 1; off < 16; off <<= 1)
          rs += __shfl_xor(rs, off, 32);
        lrun[mt][r] = lrun[mt][r] * scal + rs;
        mrun[mt][r] = mnew;
#pragma unroll
        for (int s = 0; s < 4; ++s) oacc[mt][s][r] *= scal;
        p0a[r] = p0; p1a[r] = p1;
      }
      _Float16* ps = &Ps[wave][0];
#pragma unroll
      for (int r = 0; r < 8; ++r) {
        ps[(r + 8 * half) * 32 + ln]      = (_Float16)p0a[r];
        ps[(r + 8 * half) * 32 + 16 + ln] = (_Float16)p1a[r];
      }
      v16h pa = load_a16(ps, 32, 0, 0);
#pragma unroll
      for (int s = 0; s < 4; ++s) {
        v16h b = *(const v16h*)(Vt + (s * 16 + ln) * 32 + 16 * half);
        oacc[mt][s] = wmma32(pa, b, oacc[mt][s]);
      }
    }
    __syncthreads();
  }

  // normalize + write pixel-major: token (gidx) -> channel c, pixel base p0
#pragma unroll
  for (int mt = 0; mt < 2; ++mt)
#pragma unroll
    for (int s = 0; s < 4; ++s)
#pragma unroll
      for (int r = 0; r < 8; ++r) {
        int R    = wave * 32 + mt * 16 + r + 8 * half;
        int gidx = blockIdx.x * 256 + R;
        int ch   = gidx / 576;
        int p    = (gidx % 576) * 64 + s * 16 + ln;
        float val = oacc[mt][s][r] / lrun[mt][r];
        out_pm[(size_t)p * INNERD + ch] = (_Float16)val;
      }
}

// ============================ host orchestration =============================
extern "C" void kernel_launch(void* const* d_in, const int* in_sizes, int n_in,
                              void* d_out, int out_size, void* d_ws, size_t ws_size,
                              hipStream_t stream) {
  (void)in_sizes; (void)n_in; (void)out_size; (void)ws_size;
  const int N = NPIX;
  float* x = (float*)d_out;  // residual stream (f32 [128][36864])

  char* ws = (char*)d_ws;
  const size_t XN_B  = (size_t)CDIM * N * sizeof(_Float16);    //  9.4 MB
  const size_t BIG_B = (size_t)INNERD * N * sizeof(_Float16);  // 37.7 MB
  _Float16* xn_pm = (_Float16*)ws;                         // [N][128]
  _Float16* qb    = (_Float16*)(ws + XN_B);                // c-major / h1_pm
  _Float16* kb    = (_Float16*)(ws + XN_B + BIG_B);        // c-major / h2_pm
  _Float16* vb    = (_Float16*)(ws + XN_B + 2 * BIG_B);    // c-major (contig after kb)
  _Float16* ob    = (_Float16*)(ws + XN_B + 3 * BIG_B);    // attn out, pixel-major
  _Float16* wgt   = (_Float16*)(ws + XN_B + 4 * BIG_B);    // f16 weights

  // per-layer f16 weight offsets (halves)
  const size_t WQ_O = 0, WKV_O = 65536, WO_O = 196608, W1_O = 262144,
               W3_O = 327680, W2_O = 393216, LW = 2752512;

  // ---- weight conversion (deterministic, every call) ----
  for (int l = 0; l < 2; ++l) {
    _Float16* base = wgt + (size_t)l * LW;
    cvt_f16<<<256, 256, 0, stream>>>((const float*)d_in[3] + (size_t)l * 65536,
                                     base + WQ_O, 65536);
    cvt_f16<<<512, 256, 0, stream>>>((const float*)d_in[4] + (size_t)l * 131072,
                                     base + WKV_O, 131072);
    cvt_f16<<<256, 256, 0, stream>>>((const float*)d_in[5] + (size_t)l * 65536,
                                     base + WO_O, 65536);
    cvt_f16<<<256, 256, 0, stream>>>((const float*)d_in[8] + (size_t)l * 65536,
                                     base + W1_O, 65536);
    cvt_f16<<<256, 256, 0, stream>>>((const float*)d_in[12] + (size_t)l * 65536,
                                     base + W3_O, 65536);
    cvt_w2<<<9216, 256, 0, stream>>>((const float*)d_in[10] + (size_t)l * 2359296,
                                     base + W2_O);
  }

  hipMemcpyAsync(x, d_in[0], sizeof(float) * CDIM * N, hipMemcpyDeviceToDevice,
                 stream);

  for (int l = 0; l < 2; ++l) {
    const float* ln1g = (const float*)d_in[1] + (size_t)l * CDIM;
    const float* ln1b = (const float*)d_in[2] + (size_t)l * CDIM;
    const float* ln2g = (const float*)d_in[6] + (size_t)l * CDIM;
    const float* ln2b = (const float*)d_in[7] + (size_t)l * CDIM;
    const float* b1   = (const float*)d_in[9] + (size_t)l * INNERD;
    const float* b2   = (const float*)d_in[11] + (size_t)l * INNERD;
    const float* b3   = (const float*)d_in[13] + (size_t)l * CDIM;
    _Float16* wb = wgt + (size_t)l * LW;

    // ---- attention branch ----
    ln_kernel<<<144, 256, 0, stream>>>(x, ln1g, ln1b, xn_pm);
    gemm_wmma<<<dim3(N / 64, 4), 128, 0, stream>>>(wb + WQ_O, xn_pm, nullptr,
        nullptr, qb, nullptr, 512, 128, N, 1);
    gemm_wmma<<<dim3(N / 64, 8), 128, 0, stream>>>(wb + WKV_O, xn_pm, nullptr,
        nullptr, kb, nullptr, 1024, 128, N, 1);   // k rows then v rows
    rotary_kernel<<<(INNERD * N) / 256, 256, 0, stream>>>(qb, kb);
    attn_kernel<<<1152, 256, 0, stream>>>(qb, kb, vb, ob);
    gemm_wmma<<<dim3(N / 64, 1), 128, 0, stream>>>(wb + WO_O, ob, nullptr,
        nullptr, nullptr, x, 128, 512, N, 2);     // x += wo @ attn_out

    // ---- LeFF branch ----
    ln_kernel<<<144, 256, 0, stream>>>(x, ln2g, ln2b, xn_pm);
    gemm_wmma<<<dim3(N / 64, 4), 128, 0, stream>>>(wb + W1_O, xn_pm, b1,
        qb, nullptr, nullptr, 512, 128, N, 0);    // h1 pixel-major (reuses qb)
    conv3x3_gelu<<<dim3(N / 64, 4), 128, 0, stream>>>(wb + W2_O, qb, b2, kb);
    gemm_wmma<<<dim3(N / 64, 1), 128, 0, stream>>>(wb + W3_O, kb, b3,
        nullptr, nullptr, x, 128, 512, N, 2);     // x += w3 @ h2
  }
}
